// ParallelTransportAttention_65635690217751
// MI455X (gfx1250) — matrix operand order repaired and checked by reading, not compiled, not run on hardware
//
#include <hip/hip_runtime.h>
#include <math.h>

// Problem constants (match reference)
#define BATCHN   2
#define SSEQ     256
#define DBASE    512
#define DFIB     32
#define NHEADS   8
#define DHEAD    64
#define LIERANK  8

typedef __attribute__((ext_vector_type(16))) _Float16 v16h;
typedef __attribute__((ext_vector_type(8)))  _Float16 v8h;
typedef __attribute__((ext_vector_type(8)))  float    v8f;

// ---------------------------------------------------------------------------
// WMMA operand loaders (gfx1250 wave32 16x16x32 f16 layouts, per CDNA5 ISA
// 7.12.2):
//  A (16x32, MxK): lane<16 -> row=lane, k in {0..7,16..23}; lane>=16 ->
//  row=lane-16, k in {8..15,24..31}. B (32x16, KxN): lane<16 -> n=lane,
//  k in {0..7,16..23}; lane>=16 -> n=lane-16, k in {8..15,24..31}.
// Both loaders issue only contiguous 128-bit reads.
// ---------------------------------------------------------------------------
static __device__ __forceinline__ v16h load_a_tile(const _Float16* A, int lda,
                                                   int row0, int kb, int lane) {
  int r  = row0 + (lane & 15);
  int k0 = kb + ((lane & 16) ? 8 : 0);
  const _Float16* p = A + (size_t)r * lda + k0;
  v8h lo = *(const v8h*)p;         // k0..k0+7
  v8h hi = *(const v8h*)(p + 16);  // k0+16..k0+23
  v16h out;
#pragma unroll
  for (int h = 0; h < 8; ++h) { out[h] = lo[h]; out[h + 8] = hi[h]; }
  return out;
}

// B operand with B stored as [N x K] row-major (computing A @ B^T).
static __device__ __forceinline__ v16h load_b_tile_t(const _Float16* Bm, int ldb,
                                                     int kb, int col0, int lane) {
  int n  = col0 + (lane & 15);
  int k0 = kb + ((lane & 16) ? 8 : 0);
  const _Float16* p = Bm + (size_t)n * ldb + k0;
  v8h lo = *(const v8h*)p;
  v8h hi = *(const v8h*)(p + 16);
  v16h out;
#pragma unroll
  for (int h = 0; h < 8; ++h) { out[h] = lo[h]; out[h + 8] = hi[h]; }
  return out;
}

// ---------------------------------------------------------------------------
// Batched WMMA GEMM: C = alpha * A * B^T (+bias), B stored [N x K].
// 128 threads = 4 waves; each wave owns a 32x32 tile (2x2 WMMAs, K-step 32).
// Batch index bz -> (b1 = bz/nB2, b2 = bz%nB2) with independent strides.
// f32 output (Cf, ldc) and/or f16 output (Ch, ldch, optionally transposed).
// ---------------------------------------------------------------------------
__global__ __launch_bounds__(128) void wmma_gemm_kernel(
    const _Float16* __restrict__ A, long sA1, long sA2,
    const _Float16* __restrict__ Bm, long sB1, long sB2,
    float* __restrict__ Cf, _Float16* __restrict__ Ch, long sC1, long sC2,
    const float* __restrict__ bias,
    int lda, int ldb, int ldc, int ldch, int transCh,
    int K, int nB2, float alpha)
{
  int lane = threadIdx.x & 31;
  int wave = threadIdx.x >> 5;
  int wm = wave >> 1, wn = wave & 1;
  int tileRow = blockIdx.x * 64 + wm * 32;
  int tileCol = blockIdx.y * 64 + wn * 32;
  int bz = blockIdx.z;
  int b1 = bz / nB2, b2 = bz - b1 * nB2;
  A  += (size_t)b1 * sA1 + (size_t)b2 * sA2;
  Bm += (size_t)b1 * sB1 + (size_t)b2 * sB2;
  size_t coff = (size_t)b1 * sC1 + (size_t)b2 * sC2;

  v8f acc00 = {}, acc01 = {}, acc10 = {}, acc11 = {};
  for (int kb = 0; kb < K; kb += 32) {
    v16h a0  = load_a_tile(A, lda, tileRow,      kb, lane);
    v16h a1  = load_a_tile(A, lda, tileRow + 16, kb, lane);
    v16h b0  = load_b_tile_t(Bm, ldb, kb, tileCol,      lane);
    v16h b1v = load_b_tile_t(Bm, ldb, kb, tileCol + 16, lane);
    acc00 = __builtin_amdgcn_wmma_f32_16x16x32_f16(false, a0, false, b0,  (short)0, acc00, false, false);
    acc01 = __builtin_amdgcn_wmma_f32_16x16x32_f16(false, a0, false, b1v, (short)0, acc01, false, false);
    acc10 = __builtin_amdgcn_wmma_f32_16x16x32_f16(false, a1, false, b0,  (short)0, acc10, false, false);
    acc11 = __builtin_amdgcn_wmma_f32_16x16x32_f16(false, a1, false, b1v, (short)0, acc11, false, false);
  }

  // C/D layout: VGPR r, lane<16 -> (m=r, n=lane); lane>=16 -> (m=8+r, n=lane-16)
  int mloc = (lane & 16) ? 8 : 0;
  int nloc = lane & 15;
#pragma unroll
  for (int r = 0; r < 8; ++r) {
    int m0 = tileRow + mloc + r;
    int m1 = m0 + 16;
    int n0 = tileCol + nloc;
    int n1 = n0 + 16;
    float bz0 = bias ? bias[n0] : 0.f;
    float bz1 = bias ? bias[n1] : 0.f;
    float v00 = acc00[r] * alpha + bz0;
    float v01 = acc01[r] * alpha + bz1;
    float v10 = acc10[r] * alpha + bz0;
    float v11 = acc11[r] * alpha + bz1;
    if (Cf) {
      Cf[coff + (size_t)m0 * ldc + n0] = v00;
      Cf[coff + (size_t)m0 * ldc + n1] = v01;
      Cf[coff + (size_t)m1 * ldc + n0] = v10;
      Cf[coff + (size_t)m1 * ldc + n1] = v11;
    }
    if (Ch) {
      if (transCh) {
        Ch[coff + (size_t)n0 * ldch + m0] = (_Float16)v00;
        Ch[coff + (size_t)n1 * ldch + m0] = (_Float16)v01;
        Ch[coff + (size_t)n0 * ldch + m1] = (_Float16)v10;
        Ch[coff + (size_t)n1 * ldch + m1] = (_Float16)v11;
      } else {
        Ch[coff + (size_t)m0 * ldch + n0] = (_Float16)v00;
        Ch[coff + (size_t)m0 * ldch + n1] = (_Float16)v01;
        Ch[coff + (size_t)m1 * ldch + n0] = (_Float16)v10;
        Ch[coff + (size_t)m1 * ldch + n1] = (_Float16)v11;
      }
    }
  }
}

// ---------------------------------------------------------------------------
// 32x32 LDS matmul helpers (block of 256 threads, 4 elems/thread)
// ---------------------------------------------------------------------------
static __device__ __forceinline__ void mm32(float* __restrict__ D,
                                            const float* X, const float* Y, int tid) {
  __syncthreads();
  for (int e = tid; e < 1024; e += 256) {
    int r = e >> 5, c = e & 31;
    float s = 0.f;
#pragma unroll
    for (int k = 0; k < 32; ++k) s = fmaf(X[r * 32 + k], Y[k * 32 + c], s);
    D[e] = s;
  }
  __syncthreads();
}

static __device__ __forceinline__ void mm32_tn(float* __restrict__ D,
                                               const float* X, const float* Y, int tid) {
  // D = X^T * Y
  __syncthreads();
  for (int e = tid; e < 1024; e += 256) {
    int r = e >> 5, c = e & 31;
    float s = 0.f;
#pragma unroll
    for (int k = 0; k < 32; ++k) s = fmaf(X[k * 32 + r], Y[k * 32 + c], s);
    D[e] = s;
  }
  __syncthreads();
}

// ---------------------------------------------------------------------------
// K1: A = 0.1*sum_r c_r G_r ; T = expm(A) via scaling(1/32)+Taylor(10)+5
// squarings; M = T^T T stored as f16 [B,S,32,32]. One block per (z,s).
// ---------------------------------------------------------------------------
__global__ __launch_bounds__(256) void expm_gram_kernel(
    const float* __restrict__ conn, const float* __restrict__ gens,
    _Float16* __restrict__ Mh)
{
  __shared__ float As[1024], P[1024], E[1024], T[1024];
  __shared__ float c[LIERANK];
  int tid = threadIdx.x;
  size_t zs = blockIdx.x;
  if (tid < LIERANK) c[tid] = conn[zs * LIERANK + tid];
  __syncthreads();
  for (int e = tid; e < 1024; e += 256) {
    float a = 0.f;
#pragma unroll
    for (int r = 0; r < LIERANK; ++r) a = fmaf(c[r], gens[r * 1024 + e], a);
    a *= 0.1f * (1.f / 32.f);
    As[e] = a;
    E[e] = (((e >> 5) == (e & 31)) ? 1.f : 0.f) + a;
    P[e] = a;
  }
  for (int t = 2; t <= 10; ++t) {
    mm32(T, P, As, tid);
    float inv = 1.f / (float)t;
    for (int e = tid; e < 1024; e += 256) { float pn = T[e] * inv; P[e] = pn; E[e] += pn; }
  }
  for (int q = 0; q < 5; ++q) {
    mm32(T, E, E, tid);
    for (int e = tid; e < 1024; e += 256) E[e] = T[e];
  }
  mm32_tn(T, E, E, tid);  // M = T^T T (symmetric)
  for (int e = tid; e < 1024; e += 256) Mh[zs * 1024 + e] = (_Float16)T[e];
}

// ---------------------------------------------------------------------------
// K2: holonomy. hol^2[i,j] = tr((M_i M_j)^2) - 2 tr(M_i M_j) + 32.
// One wave per (i,j) pair with j>=i (hol is symmetric). P = M_i @ M_j via
// 4x WMMA 16x16x32 f16 (K=32 covers full contraction). M_j symmetric ->
// B operand loaded with contiguous 128-bit reads. tr and tr(P^2) via LDS.
// NOTE: no __syncthreads() here (waves early-exit); LDS traffic is
// wave-private and DS ops from one wave execute in order.
// ---------------------------------------------------------------------------
#define HOLW 8
__global__ __launch_bounds__(256) void holonomy_kernel(
    const _Float16* __restrict__ Mh, float* __restrict__ hol)
{
  __shared__ float Pl[HOLW][1024];
  int lane = threadIdx.x & 31;
  int w = threadIdx.x >> 5;
  int z = blockIdx.y;
  int flat = blockIdx.x * HOLW + w;          // over S*S
  int i = flat >> 8;
  int j = flat & 255;
  float* hrow = hol + ((size_t)z * SSEQ + i) * SSEQ;
  if (j < i) return;                          // symmetry (wave-uniform)
  if (j == i) { if (lane == 0) hrow[j] = 0.f; return; }

  const _Float16* Mi = Mh + ((size_t)z * SSEQ + i) * 1024;
  const _Float16* Mj = Mh + ((size_t)z * SSEQ + j) * 1024;

  v16h a0 = load_a_tile(Mi, 32, 0,  0, lane);
  v16h a1 = load_a_tile(Mi, 32, 16, 0, lane);
  v16h b0 = load_b_tile_t(Mj, 32, 0, 0,  lane);   // M_j symmetric
  v16h b1 = load_b_tile_t(Mj, 32, 0, 16, lane);

  v8f p00 = {}, p01 = {}, p10 = {}, p11 = {};
  p00 = __builtin_amdgcn_wmma_f32_16x16x32_f16(false, a0, false, b0, (short)0, p00, false, false);
  p01 = __builtin_amdgcn_wmma_f32_16x16x32_f16(false, a0, false, b1, (short)0, p01, false, false);
  p10 = __builtin_amdgcn_wmma_f32_16x16x32_f16(false, a1, false, b0, (short)0, p10, false, false);
  p11 = __builtin_amdgcn_wmma_f32_16x16x32_f16(false, a1, false, b1, (short)0, p11, false, false);

  int mloc = (lane & 16) ? 8 : 0;
  int nloc = lane & 15;
#pragma unroll
  for (int r = 0; r < 8; ++r) {
    Pl[w][(mloc + r) * 32 + nloc]            = p00[r];
    Pl[w][(mloc + r) * 32 + 16 + nloc]       = p01[r];
    Pl[w][(16 + mloc + r) * 32 + nloc]       = p10[r];
    Pl[w][(16 + mloc + r) * 32 + 16 + nloc]  = p11[r];
  }
  // Compiler-level ordering barrier: keeps the DS stores above before the DS
  // loads below; hardware keeps same-wave LDS ops in order and the compiler
  // emits the required s_wait_dscnt for load results.
  asm volatile("" ::: "memory");
  float s1 = Pl[w][lane * 32 + lane];
  float s2 = 0.f;
#pragma unroll
  for (int t = 0; t < 32; ++t)
    s2 = fmaf(Pl[w][lane * 32 + t], Pl[w][t * 32 + lane], s2);
#pragma unroll
  for (int o = 16; o > 0; o >>= 1) {
    s1 += __shfl_xor(s1, o);
    s2 += __shfl_xor(s2, o);
  }
  if (lane == 0) {
    float h2 = s2 - 2.f * s1 + (float)DFIB;
    float hv = sqrtf(fmaxf(h2, 0.f));
    hrow[j] = hv;
    hol[((size_t)z * SSEQ + j) * SSEQ + i] = hv;  // mirror
  }
}

// ---------------------------------------------------------------------------
// K3: softmax row kernel. One wave per (z,h,i) row of 256; subtract
// lambda*hol, softmax, emit attn f16, accumulate total_hol via atomics.
// ---------------------------------------------------------------------------
__global__ __launch_bounds__(256) void softmax_kernel(
    const float* __restrict__ scores, const float* __restrict__ hol,
    const float* __restrict__ lambda_p, _Float16* __restrict__ attn_h,
    float* __restrict__ total_hol)
{
  int lane = threadIdx.x & 31;
  int w = threadIdx.x >> 5;
  size_t row = (size_t)blockIdx.x * 8 + w;   // over B*H*S = 4096
  int z = (int)(row >> 11);
  int i = (int)(row & 255);
  const float* sr = scores + row * SSEQ;
  const float* hr = hol + ((size_t)z * SSEQ + i) * SSEQ;
  float lam = *lambda_p;

  float v[8], hv[8];
  float mx = -3.4e38f;
#pragma unroll
  for (int t = 0; t < 8; ++t) {
    int j = lane + 32 * t;
    hv[t] = hr[j];
    v[t] = sr[j] - lam * hv[t];
    mx = fmaxf(mx, v[t]);
  }
#pragma unroll
  for (int o = 16; o > 0; o >>= 1) mx = fmaxf(mx, __shfl_xor(mx, o));
  float sum = 0.f;
#pragma unroll
  for (int t = 0; t < 8; ++t) { v[t] = __expf(v[t] - mx); sum += v[t]; }
#pragma unroll
  for (int o = 16; o > 0; o >>= 1) sum += __shfl_xor(sum, o);
  float inv = 1.f / sum;
  float th = 0.f;
#pragma unroll
  for (int t = 0; t < 8; ++t) {
    float a = v[t] * inv;
    attn_h[row * SSEQ + lane + 32 * t] = (_Float16)a;
    th = fmaf(a, hv[t], th);
  }
#pragma unroll
  for (int o = 16; o > 0; o >>= 1) th += __shfl_xor(th, o);
  if (lane == 0) atomicAdd(total_hol, th);
}

// K4: avg over heads -> f16 [B,S,S]
__global__ __launch_bounds__(256) void avg_attn_kernel(
    const _Float16* __restrict__ attn_h, _Float16* __restrict__ avg_h)
{
  size_t idx = (size_t)blockIdx.x * 256 + threadIdx.x;  // B*S*S = 131072
  int z = (int)(idx >> 16);
  size_t rem = idx & 65535;
  float s = 0.f;
#pragma unroll
  for (int h = 0; h < NHEADS; ++h)
    s += (float)attn_h[(((size_t)z * NHEADS + h) << 16) + rem];
  avg_h[idx] = (_Float16)(s * (1.f / NHEADS));
}

// ---------------------------------------------------------------------------
// K5: polar projection (replaces U @ Vh from SVD) via Newton-Schulz:
// Y0 = X/||X||_F, Y <- 1.5 Y - 0.5 Y (Y^T Y). One block per (z,s).
// ---------------------------------------------------------------------------
__global__ __launch_bounds__(256) void polar_kernel(
    const float* __restrict__ Xin, float* __restrict__ Yout)
{
  __shared__ float Y[1024], T1[1024], T2[1024];
  __shared__ float red[256];
  int tid = threadIdx.x;
  size_t zs = blockIdx.x;
  float part = 0.f;
  for (int e = tid; e < 1024; e += 256) {
    float x = Xin[zs * 1024 + e];
    Y[e] = x;
    part = fmaf(x, x, part);
  }
  red[tid] = part;
  __syncthreads();
  for (int s = 128; s > 0; s >>= 1) {
    if (tid < s) red[tid] += red[tid + s];
    __syncthreads();
  }
  float inv = rsqrtf(red[0] + 1e-30f);
  for (int e = tid; e < 1024; e += 256) Y[e] *= inv;
  for (int it = 0; it < 15; ++it) {
    mm32_tn(T1, Y, Y, tid);   // T1 = Y^T Y
    mm32(T2, Y, T1, tid);     // T2 = Y T1
    for (int e = tid; e < 1024; e += 256) Y[e] = 1.5f * Y[e] - 0.5f * T2[e];
  }
  __syncthreads();
  for (int e = tid; e < 1024; e += 256) Yout[zs * 1024 + e] = Y[e];
}

// f32 -> f16 conversion (row-major copy)
__global__ __launch_bounds__(256) void cvt_f16_kernel(
    const float* __restrict__ in, _Float16* __restrict__ out, long n)
{
  long i = (long)blockIdx.x * 256 + threadIdx.x;
  if (i < n) out[i] = (_Float16)in[i];
}

// f32 [R x C] -> f16 [C x R] tiled transpose-convert (32x32 tiles via LDS)
__global__ __launch_bounds__(256) void cvt_tr_kernel(
    const float* __restrict__ in, _Float16* __restrict__ out, int R, int C)
{
  __shared__ _Float16 tile[32][33];
  int tx = threadIdx.x & 31, ty = threadIdx.x >> 5;   // 32 x 8
  int r0 = blockIdx.y * 32, c0 = blockIdx.x * 32;
#pragma unroll
  for (int rr = ty; rr < 32; rr += 8)
    tile[rr][tx] = (_Float16)in[(size_t)(r0 + rr) * C + c0 + tx];
  __syncthreads();
#pragma unroll
  for (int rr = ty; rr < 32; rr += 8)
    out[(size_t)(c0 + rr) * R + r0 + tx] = tile[tx][rr];
}

// ---------------------------------------------------------------------------
// Orchestration
// ---------------------------------------------------------------------------
extern "C" void kernel_launch(void* const* d_in, const int* in_sizes, int n_in,
                              void* d_out, int out_size, void* d_ws, size_t ws_size,
                              hipStream_t stream)
{
  (void)in_sizes; (void)n_in; (void)out_size; (void)ws_size;
  const float* base  = (const float*)d_in[0];
  const float* fiber = (const float*)d_in[1];
  const float* conn  = (const float*)d_in[2];
  const float* gens  = (const float*)d_in[3];
  const float* Wq = (const float*)d_in[4];
  const float* bq = (const float*)d_in[5];
  const float* Wk = (const float*)d_in[6];
  const float* bk = (const float*)d_in[7];
  const float* Wv = (const float*)d_in[8];
  const float* bv = (const float*)d_in[9];
  const float* Wo = (const float*)d_in[10];
  const float* bo = (const float*)d_in[11];
  const float* lam = (const float*)d_in[12];

  float* out_attn  = (float*)d_out;                 // [2,256,512]
  float* out_fiber = out_attn + 262144;             // [2,256,32,32]
  float* out_hol   = out_attn + 262144 + 524288;    // scalar

  char* ws = (char*)d_ws;
  size_t off = 0;
  auto carve = [&](size_t bytes) -> char* {
    char* p = ws + off;
    off += (bytes + 255) & ~(size_t)255;
    return p;
  };
  _Float16* base_h  = (_Float16*)carve(512 * 512 * 2);   // [B*S, 512] row-major
  _Float16* Wq_t    = (_Float16*)carve(512 * 512 * 2);   // [N, K]
  _Float16* Wk_t    = (_Float16*)carve(512 * 512 * 2);
  _Float16* Wv_t    = (_Float16*)carve(512 * 512 * 2);
  _Float16* Wo_t    = (_Float16*)carve(512 * 512 * 2);
  _Float16* Qh      = (_Float16*)carve(512 * 512 * 2);   // [B*S, 512]
  _Float16* Kh      = (_Float16*)carve(512 * 512 * 2);   // [B*S, 512]
  _Float16* Vt_h    = (_Float16*)carve(512 * 512 * 2);   // [512, B*S] (transposed)
  _Float16* Mh      = (_Float16*)carve((size_t)BATCHN * SSEQ * 1024 * 2);
  float*    holbuf  = (float*)   carve((size_t)BATCHN * SSEQ * SSEQ * 4);
  float*    scores  = (float*)   carve((size_t)BATCHN * NHEADS * SSEQ * SSEQ * 4);
  _Float16* attn_h  = (_Float16*)carve((size_t)BATCHN * NHEADS * SSEQ * SSEQ * 2);
  _Float16* attnV_h = (_Float16*)carve(512 * 512 * 2);
  _Float16* avg_h   = (_Float16*)carve((size_t)BATCHN * SSEQ * SSEQ * 2);
  _Float16* fiber_t = (_Float16*)carve((size_t)BATCHN * SSEQ * 1024 * 2); // per z: [1024, 256]
  float*    nf_raw  = (float*)   carve((size_t)BATCHN * SSEQ * 1024 * 4);

  (void)hipMemsetAsync(out_hol, 0, sizeof(float), stream);

  // conversions: base row-major, weights transposed to [N,K], fiber to [z][1024][256]
  cvt_f16_kernel<<<1024, 256, 0, stream>>>(base, base_h, 262144);
  cvt_tr_kernel<<<dim3(16, 16), 256, 0, stream>>>(Wq, Wq_t, 512, 512);
  cvt_tr_kernel<<<dim3(16, 16), 256, 0, stream>>>(Wk, Wk_t, 512, 512);
  cvt_tr_kernel<<<dim3(16, 16), 256, 0, stream>>>(Wv, Wv_t, 512, 512);
  cvt_tr_kernel<<<dim3(16, 16), 256, 0, stream>>>(Wo, Wo_t, 512, 512);
  cvt_tr_kernel<<<dim3(32, 8), 256, 0, stream>>>(fiber,          fiber_t,          256, 1024);
  cvt_tr_kernel<<<dim3(32, 8), 256, 0, stream>>>(fiber + 262144, fiber_t + 262144, 256, 1024);

  // expm -> gram M, then pairwise holonomy
  expm_gram_kernel<<<BATCHN * SSEQ, 256, 0, stream>>>(conn, gens, Mh);
  holonomy_kernel<<<dim3(SSEQ * SSEQ / HOLW, BATCHN, 1), 256, 0, stream>>>(Mh, holbuf);

  // Q,K projections (row-major f16 out); V projection stored transposed [512, B*S]
  wmma_gemm_kernel<<<dim3(8, 8, 1), 128, 0, stream>>>(
      base_h, 0, 0, Wq_t, 0, 0, nullptr, Qh, 0, 0, bq,
      512, 512, 512, 512, 0, 512, 1, 1.0f);
  wmma_gemm_kernel<<<dim3(8, 8, 1), 128, 0, stream>>>(
      base_h, 0, 0, Wk_t, 0, 0, nullptr, Kh, 0, 0, bk,
      512, 512, 512, 512, 0, 512, 1, 1.0f);
  wmma_gemm_kernel<<<dim3(8, 8, 1), 128, 0, stream>>>(
      base_h, 0, 0, Wv_t, 0, 0, nullptr, Vt_h, 0, 0, bv,
      512, 512, 512, 512, 1, 512, 1, 1.0f);

  // scores = Q K^T / 8 per (z,h): M=N=256, K=64, batch=16 (K row-major == B^T form)
  wmma_gemm_kernel<<<dim3(4, 4, 16), 128, 0, stream>>>(
      Qh, 131072, 64, Kh, 131072, 64, scores, nullptr, 524288, 65536, nullptr,
      512, 512, 256, 256, 0, 64, NHEADS, 0.125f);

  // softmax(+hol bias) and total_hol
  softmax_kernel<<<512, 256, 0, stream>>>(scores, holbuf, lam, attn_h, out_hol);

  // attn @ V per (z,h): M=256, N=64, K=256; V supplied transposed [512, B*S]
  wmma_gemm_kernel<<<dim3(4, 1, 16), 128, 0, stream>>>(
      attn_h, 524288, 65536, Vt_h, 256, 32768, nullptr, attnV_h, 131072, 64, nullptr,
      256, 512, 512, 512, 0, 256, NHEADS, 1.0f);

  // output projection -> d_out part 1
  wmma_gemm_kernel<<<dim3(8, 8, 1), 128, 0, stream>>>(
      attnV_h, 0, 0, Wo_t, 0, 0, out_attn, nullptr, 0, 0, bo,
      512, 512, 512, 512, 0, 512, 1, 1.0f);

  // avg over heads, fiber update GEMM: [256,256] @ [256,1024] per z (fiber_t = B^T)
  avg_attn_kernel<<<512, 256, 0, stream>>>(attn_h, avg_h);
  wmma_gemm_kernel<<<dim3(4, 16, 2), 128, 0, stream>>>(
      avg_h, 65536, 0, fiber_t, 262144, 0, nf_raw, nullptr, 262144, 0, nullptr,
      256, 256, 1024, 1024, 0, 256, 1, 1.0f);

  // polar projection -> d_out part 2
  polar_kernel<<<BATCHN * SSEQ, 256, 0, stream>>>(nf_raw, out_fiber);
}
